// TestGatedModel_45827301048694
// MI455X (gfx1250) — 8-bit matrix operand order repaired and checked by reading, compiled, not run on hardware
//
#include <hip/hip_runtime.h>
#include <stdint.h>

#define HIDDEN 4096
#define GROUP  128
#define NBLK   (HIDDEN / GROUP)
#define FP8MAX 448.0f
#define LSTRIDE 144

typedef int   v16i __attribute__((ext_vector_type(16)));
typedef float v8f  __attribute__((ext_vector_type(8)));

__device__ __forceinline__ unsigned sw_fp8_e4m3(float f) {
  unsigned u   = __float_as_uint(f);
  unsigned sgn = (u >> 31) << 7;
  float ax = fabsf(f);
  if (ax != ax)    return sgn | 0x7Fu;
  if (ax >= 448.f) return sgn | 0x7Eu;
  int e;
  float m = frexpf(ax, &e);
  int E = e + 6;
  if (E <= 0) {
    unsigned q = (unsigned)rintf(ax * 512.f);
    return sgn | (q > 8u ? 8u : q);
  }
  unsigned q = (unsigned)rintf(m * 16.f) - 8u;
  if (q == 8u) { q = 0u; ++E; }
  if (E >= 16) return sgn | 0x7Eu;
  if (E == 15 && q == 7u) q = 6u;
  return sgn | ((unsigned)E << 3) | q;
}

__device__ __forceinline__ unsigned pack_fp8x4(float a, float b, float c, float d) {
#if __has_builtin(__builtin_amdgcn_cvt_pk_fp8_f32)
  int v = __builtin_amdgcn_cvt_pk_fp8_f32(a, b, 0, false);
  v     = __builtin_amdgcn_cvt_pk_fp8_f32(c, d, v, true);
  return (unsigned)v;
#else
  return sw_fp8_e4m3(a) | (sw_fp8_e4m3(b) << 8) |
         (sw_fp8_e4m3(c) << 16) | (sw_fp8_e4m3(d) << 24);
#endif
}

__device__ __forceinline__ void async_cp64(unsigned lds, const uint8_t* g) {
  asm volatile(
      "global_load_async_to_lds_b128 %0, %1, off\n\t"
      "global_load_async_to_lds_b128 %0, %1, off offset:16\n\t"
      "global_load_async_to_lds_b128 %0, %1, off offset:32\n\t"
      "global_load_async_to_lds_b128 %0, %1, off offset:48"
      :: "v"(lds), "v"(g) : "memory");
}

__device__ __forceinline__ void wait_async_all() {
  asm volatile("s_wait_asynccnt 0x0" ::: "memory");
}

__global__ __launch_bounds__(256) void k_prologue(
    const float* __restrict__ x, const float* __restrict__ z,
    const float* __restrict__ nw, uint8_t* __restrict__ a8,
    float* __restrict__ asc_t, int nrows)
{
  const int  wave = threadIdx.x >> 5;
  const int  lane = threadIdx.x & 31;
  const long pair = (long)blockIdx.x * 8 + wave;
  const int  head = (int)(pair & (NBLK - 1));
  const long row  = pair >> 5;
  const long base = row * HIDDEN + (long)head * GROUP;

  const float4 xv = *(const float4*)(x  + base + lane * 4);
  const float4 zv = *(const float4*)(z  + base + lane * 4);
  const float4 wv = *(const float4*)(nw + lane * 4);

  float xr[4] = {fmaxf(xv.x, 0.f), fmaxf(xv.y, 0.f), fmaxf(xv.z, 0.f), fmaxf(xv.w, 0.f)};
  float zr[4] = {fmaxf(zv.x, 0.f), fmaxf(zv.y, 0.f), fmaxf(zv.z, 0.f), fmaxf(zv.w, 0.f)};

  float ss = xr[0]*xr[0] + xr[1]*xr[1] + xr[2]*xr[2] + xr[3]*xr[3];
  #pragma unroll
  for (int i = 1; i < 32; i <<= 1) ss += __shfl_xor(ss, i, 32);
  const float rs = rsqrtf(ss * (1.f / 128.f) + 1e-6f);

  const float wl[4] = {wv.x, wv.y, wv.z, wv.w};
  float g[4];
  #pragma unroll
  for (int j = 0; j < 4; ++j) {
    const float silu = zr[j] / (1.f + __expf(-zr[j]));
    g[j] = xr[j] * rs * wl[j] * silu;
  }

  float am = fmaxf(fmaxf(fabsf(g[0]), fabsf(g[1])), fmaxf(fabsf(g[2]), fabsf(g[3])));
  #pragma unroll
  for (int i = 1; i < 32; i <<= 1) am = fmaxf(am, __shfl_xor(am, i, 32));

  const float sc  = fmaxf(am, 1e-12f) * (1.f / FP8MAX);
  const float inv = 1.f / sc;
  const unsigned packed = pack_fp8x4(g[0]*inv, g[1]*inv, g[2]*inv, g[3]*inv);
  ((unsigned*)(a8 + base))[lane] = packed;
  if (lane == 0) asc_t[(long)head * nrows + row] = sc;
}

__global__ __launch_bounds__(256) void k_wquant(
    const float* __restrict__ w, uint8_t* __restrict__ w8,
    float* __restrict__ wsc)
{
  __shared__ float red[8];
  const int nb = blockIdx.x, kb = blockIdx.y;
  const int t  = threadIdx.x;
  const int r  = t >> 1;
  const int ch = (t & 1) * 64;
  const long rowoff = (long)(nb * GROUP + r) * HIDDEN + kb * GROUP + ch;

  float4 v[16];
  float am = 0.f;
  #pragma unroll
  for (int i = 0; i < 16; ++i) {
    v[i] = *(const float4*)(w + rowoff + i * 4);
    am = fmaxf(am, fmaxf(fmaxf(fabsf(v[i].x), fabsf(v[i].y)),
                         fmaxf(fabsf(v[i].z), fabsf(v[i].w))));
  }
  #pragma unroll
  for (int i = 1; i < 32; i <<= 1) am = fmaxf(am, __shfl_xor(am, i, 32));
  if ((t & 31) == 0) red[t >> 5] = am;
  __syncthreads();
  float bm = red[0];
  #pragma unroll
  for (int i = 1; i < 8; ++i) bm = fmaxf(bm, red[i]);

  const float sc  = fmaxf(bm, 1e-12f) * (1.f / FP8MAX);
  const float inv = 1.f / sc;
  unsigned* op = (unsigned*)(w8 + rowoff);
  #pragma unroll
  for (int i = 0; i < 16; ++i)
    op[i] = pack_fp8x4(v[i].x*inv, v[i].y*inv, v[i].z*inv, v[i].w*inv);
  if (t == 0) wsc[nb * NBLK + kb] = sc;
}

__global__ __launch_bounds__(256, 1) void k_gemm(
    const uint8_t* __restrict__ a8, const uint8_t* __restrict__ w8,
    const float* __restrict__ a_st, const float* __restrict__ w_s,
    float* __restrict__ out, int nrows)
{
  __shared__ uint8_t sA[2][128 * LSTRIDE];
  __shared__ uint8_t sB[2][128 * LSTRIDE];

  const int t    = threadIdx.x;
  const int lane = t & 31;
  const int wave = t >> 5;
  const int wm = wave & 3, wn = wave >> 2;
  const int m0 = blockIdx.x * 128 + wm * 32;
  const int n0 = blockIdx.y * 128 + wn * 64;
  const int lh = lane >> 4;
  const int ll = lane & 15;

  const uint8_t* gA = a8 + (long)blockIdx.x * 128 * HIDDEN;
  const uint8_t* gB = w8 + (long)blockIdx.y * 128 * HIDDEN;
  const int srow = t >> 1;
  const int shalf = (t & 1) * 64;

  v8f acc[2][4];
  #pragma unroll
  for (int sm = 0; sm < 2; ++sm)
    #pragma unroll
    for (int sn = 0; sn < 4; ++sn)
      acc[sm][sn] = (v8f){0.f,0.f,0.f,0.f,0.f,0.f,0.f,0.f};

  async_cp64((unsigned)(uintptr_t)(&sA[0][srow * LSTRIDE + shalf]),
             gA + (long)srow * HIDDEN + shalf);
  async_cp64((unsigned)(uintptr_t)(&sB[0][srow * LSTRIDE + shalf]),
             gB + (long)srow * HIDDEN + shalf);
  wait_async_all();
  __syncthreads();

  int buf = 0;
  for (int kb = 0; kb < NBLK; ++kb) {
    const int k0 = kb * GROUP;

    if (kb + 1 < NBLK) {
      async_cp64((unsigned)(uintptr_t)(&sA[buf ^ 1][srow * LSTRIDE + shalf]),
                 gA + (long)srow * HIDDEN + k0 + GROUP + shalf);
      async_cp64((unsigned)(uintptr_t)(&sB[buf ^ 1][srow * LSTRIDE + shalf]),
                 gB + (long)srow * HIDDEN + k0 + GROUP + shalf);
    }
    if (kb + 2 < NBLK) {
      __builtin_prefetch(gA + (long)srow * HIDDEN + k0 + 2 * GROUP, 0, 3);
      __builtin_prefetch(gB + (long)srow * HIDDEN + k0 + 2 * GROUP, 0, 3);
    }

    const float wsv = w_s[blockIdx.y * NBLK + kb];

    v16i afr[2];
    #pragma unroll
    for (int sm = 0; sm < 2; ++sm) {
      const uint8_t* ap = &sA[buf][(wm * 32 + sm * 16 + ll) * LSTRIDE + 16 * lh];
      #pragma unroll
      for (int h2 = 0; h2 < 2; ++h2)
        #pragma unroll
        for (int j = 0; j < 4; ++j) {
          const int2 d = *(const int2*)(ap + h2 * 64 + 32 * (j >> 1) + 8 * (j & 1));
          afr[sm][h2 * 8 + 2 * j]     = d.x;
          afr[sm][h2 * 8 + 2 * j + 1] = d.y;
        }
    }

    v16i bfr[4];
    #pragma unroll
    for (int sn = 0; sn < 4; ++sn) {
      const uint8_t* bp = &sB[buf][(wn * 64 + sn * 16 + ll) * LSTRIDE + 16 * lh];
      #pragma unroll
      for (int q = 0; q < 4; ++q) {
        const int4 d = *(const int4*)(bp + 32 * q);
        bfr[sn][4*q] = d.x; bfr[sn][4*q+1] = d.y;
        bfr[sn][4*q+2] = d.z; bfr[sn][4*q+3] = d.w;
      }
    }

    float cs[2][8];
    #pragma unroll
    for (int sm = 0; sm < 2; ++sm) {
      const float* sp = a_st + (long)kb * nrows + m0 + sm * 16 + 8 * lh;
      const float4 s0 = *(const float4*)(sp);
      const float4 s1 = *(const float4*)(sp + 4);
      cs[sm][0] = s0.x * wsv; cs[sm][1] = s0.y * wsv;
      cs[sm][2] = s0.z * wsv; cs[sm][3] = s0.w * wsv;
      cs[sm][4] = s1.x * wsv; cs[sm][5] = s1.y * wsv;
      cs[sm][6] = s1.z * wsv; cs[sm][7] = s1.w * wsv;
    }

    #pragma unroll
    for (int sm = 0; sm < 2; ++sm)
      #pragma unroll
      for (int sn = 0; sn < 4; ++sn) {
        v8f p = (v8f){0.f,0.f,0.f,0.f,0.f,0.f,0.f,0.f};
        p = __builtin_amdgcn_wmma_f32_16x16x128_fp8_fp8(
                afr[sm], bfr[sn], (short)0, p, false, false);
        #pragma unroll
        for (int r = 0; r < 8; ++r)
          acc[sm][sn][r] += p[r] * cs[sm][r];
      }

    wait_async_all();
    __syncthreads();
    buf ^= 1;
  }

  #pragma unroll
  for (int sm = 0; sm < 2; ++sm)
    #pragma unroll
    for (int r = 0; r < 8; ++r) {
      const long m = m0 + sm * 16 + r + 8 * lh;
      #pragma unroll
      for (int sn = 0; sn < 4; ++sn)
        out[m * HIDDEN + n0 + sn * 16 + ll] = acc[sm][sn][r];
    }
}

extern "C" void kernel_launch(void* const* d_in, const int* in_sizes, int n_in,
                              void* d_out, int out_size, void* d_ws, size_t ws_size,
                              hipStream_t stream)
{
  const float* x  = (const float*)d_in[0];
  const float* z  = (const float*)d_in[1];
  const float* nw = (const float*)d_in[2];
  const float* w  = (const float*)d_in[3];
  float* out = (float*)d_out;
  const int n = in_sizes[0] / HIDDEN;

  uint8_t* ws8 = (uint8_t*)d_ws;
  size_t off = 0;
  uint8_t* a8  = ws8 + off;  off += (size_t)n * HIDDEN;
  uint8_t* w8  = ws8 + off;  off += (size_t)HIDDEN * HIDDEN;
  float*   asc = (float*)(ws8 + off); off += (size_t)n * NBLK * sizeof(float);
  float*   wsc = (float*)(ws8 + off); off += (size_t)NBLK * NBLK * sizeof(float);
  (void)ws_size; (void)n_in; (void)out_size;

  k_prologue<<<(n * NBLK) / 8, 256, 0, stream>>>(x, z, nw, a8, asc, n);
  k_wquant<<<dim3(NBLK, NBLK), 256, 0, stream>>>(w, w8, wsc);
  k_gemm<<<dim3(n / 128, HIDDEN / 128), 256, 0, stream>>>(a8, w8, asc, wsc, out, n);
}
